// Attention_89670327206161
// MI455X (gfx1250) — compile-verified, matrix operand order checked by reading, not run on hardware
//
#include <hip/hip_runtime.h>
#include <hip/hip_bf16.h>
#include <math.h>

// ---------------------------------------------------------------------------
// MI455X (gfx1250) attention: all GEMMs via v_wmma_f32_16x16x32_bf16 (wave32).
// Shapes: B=8 S=4096 D=1024 H=16 dh=64.  Dominant cost = 4 projections
// (~275 GFLOP bf16).  x is pre-converted to bf16 (64MB) so the A operand of
// the QKV GEMM is L2-resident across all 24 N-jobs per M-slice and the hot
// loop is pure b128-load -> wmma.  Softmax kept fp32 (tiny 64x64 per head).
// ---------------------------------------------------------------------------

typedef __bf16 bf16_t;
typedef __attribute__((ext_vector_type(16))) __bf16 v16bf;
typedef __attribute__((ext_vector_type(8)))  __bf16 v8bf;
typedef __attribute__((ext_vector_type(8)))  float  v8f;

#define DEV_INLINE __device__ __forceinline__

DEV_INLINE v8f vzero8() {
  v8f v;
#pragma unroll
  for (int e = 0; e < 8; ++e) v[e] = 0.0f;
  return v;
}

DEV_INLINE v8f wmma_bf16(v16bf a, v16bf b, v8f c) {
  // (neg_a, A, neg_b, B, c_mod, C, reuse_a, reuse_b)
  return __builtin_amdgcn_wmma_f32_16x16x32_bf16(false, a, false, b, (short)0, c,
                                                 false, false);
}

// 16(M/N) x 32(K) bf16 fragment from a row-major source: lane holds row
// (lane&15), K chunks [kg, kg+8) and [16+kg, 16+kg+8) with kg = (lane>>4)*8.
// (CDNA5 ISA "16-bit A-Matrix 16x32" layout; B uses the same
// per-lane pattern with rows = columns of B, i.e. row-major B^T source.)
DEV_INLINE v16bf load_frag_bf16(const bf16_t* src, int row0, int ld, int k0, int lane) {
  const int r  = row0 + (lane & 15);
  const int kg = (lane >> 4) << 3;
  const bf16_t* p = src + (size_t)r * ld + k0 + kg;
  v8bf lo = *(const v8bf*)p;
  v8bf hi = *(const v8bf*)(p + 16);
  v16bf f;
#pragma unroll
  for (int e = 0; e < 8; ++e) { f[e] = lo[e]; f[e + 8] = hi[e]; }
  return f;
}

// ---------------------------------------------------------------------------
// Vectorized fp32 -> bf16 conversion (8 elements/thread/iter; n must be /8)
// ---------------------------------------------------------------------------
__global__ void convert_f32_bf16_vec8_kernel(const float* __restrict__ src,
                                             bf16_t* __restrict__ dst, int nvec) {
  int i = blockIdx.x * blockDim.x + threadIdx.x;
  const int stride = gridDim.x * blockDim.x;
  for (; i < nvec; i += stride) {
    v8f v = *(const v8f*)(src + (size_t)i * 8);
    v8bf o;
#pragma unroll
    for (int e = 0; e < 8; ++e) o[e] = (bf16_t)v[e];
    *(v8bf*)(dst + (size_t)i * 8) = o;
  }
}

// ---------------------------------------------------------------------------
// Fused QKV projection: q/k/v[(b*16+h)][s][dh] (bf16) = xb @ W{q,k,v}^T
// Block = 256 threads (8 waves), tile 128(M) x 128(N); wave = 32x64 = 2x4 WMMA.
// blockIdx.x = weight(3) x ntile(8)  [adjacent blocks share the M-slice of xb
// so the A-operand stays hot in L2], blockIdx.y = mtile(256).
// ---------------------------------------------------------------------------
__global__ __launch_bounds__(256)
void qkv_proj_kernel(const bf16_t* __restrict__ xb,
                     const bf16_t* __restrict__ wq,
                     const bf16_t* __restrict__ wk,
                     const bf16_t* __restrict__ wv,
                     bf16_t* __restrict__ qh,
                     bf16_t* __restrict__ kh,
                     bf16_t* __restrict__ vh) {
  const int tid  = threadIdx.x;
  const int lane = tid & 31;
  const int wid  = tid >> 5;
  const int wsel = blockIdx.x >> 3;                       // 0=Q 1=K 2=V
  const int n0   = (blockIdx.x & 7) * 128 + (wid >> 2) * 64;
  const int m0   = blockIdx.y * 128 + (wid & 3) * 32;

  const bf16_t* W = (wsel == 0) ? wq : (wsel == 1) ? wk : wv;
  bf16_t*       O = (wsel == 0) ? qh : (wsel == 1) ? kh : vh;

  v8f acc[2][4];
#pragma unroll
  for (int i = 0; i < 2; ++i)
#pragma unroll
    for (int j = 0; j < 4; ++j) acc[i][j] = vzero8();

  for (int k0 = 0; k0 < 1024; k0 += 32) {
    if (k0 + 32 < 1024)  // hint next K-slab of the A operand (global_prefetch_b8)
      __builtin_prefetch(xb + (size_t)(m0 + (lane & 15)) * 1024 + k0 + 32, 0, 1);
    v16bf a0 = load_frag_bf16(xb, m0,      1024, k0, lane);
    v16bf a1 = load_frag_bf16(xb, m0 + 16, 1024, k0, lane);
    v16bf b[4];
#pragma unroll
    for (int j = 0; j < 4; ++j) b[j] = load_frag_bf16(W, n0 + j * 16, 1024, k0, lane);
#pragma unroll
    for (int j = 0; j < 4; ++j) {
      acc[0][j] = wmma_bf16(a0, b[j], acc[0][j]);
      acc[1][j] = wmma_bf16(a1, b[j], acc[1][j]);
    }
  }

  // Store split-heads bf16: C/D layout N=lane&15, M=8*(lane>>4)+e.
#pragma unroll
  for (int i = 0; i < 2; ++i) {
#pragma unroll
    for (int j = 0; j < 4; ++j) {
      const int n  = n0 + j * 16 + (lane & 15);
      const int h  = n >> 6;
      const int dh = n & 63;
#pragma unroll
      for (int e = 0; e < 8; ++e) {
        const int m  = m0 + i * 16 + ((lane >> 4) << 3) + e;
        const int bb = m >> 12;        // batch
        const int s  = m & 4095;       // sequence
        O[(((size_t)((bb << 4) + h) * 4096 + s) << 6) + dh] = (bf16_t)acc[i][j][e];
      }
    }
  }
}

// ---------------------------------------------------------------------------
// Scores + softmax: per (b,h):  Z[kk][qq] = sum_y K[y][kk] * Q[y][qq] / 8,
// softmax over qq, output bf16 A[y=kk][z=qq] (64x64).
// Q/K tiles (128 y x 64 d) are staged transposed in LDS so both WMMA operands
// load contiguously; each wave owns 2 of the 16 output tiles across all y.
// ---------------------------------------------------------------------------
__global__ __launch_bounds__(256)
void scores_softmax_kernel(const bf16_t* __restrict__ qh,
                           const bf16_t* __restrict__ kh,
                           bf16_t* __restrict__ attn) {
  __shared__ __align__(16) bf16_t Qt[64 * 128];  // [d][y]
  __shared__ __align__(16) bf16_t Kt[64 * 128];  // [d][y]
  __shared__ float Z[64 * 64];                   // [kk][qq]

  const int tid  = threadIdx.x;
  const int lane = tid & 31;
  const int wid  = tid >> 5;
  const int bh   = blockIdx.x;
  const bf16_t* Q = qh + (size_t)bh * 4096 * 64;
  const bf16_t* K = kh + (size_t)bh * 4096 * 64;

  const int ti  = wid >> 1;          // kk tile (0..3)
  const int tj0 = (wid & 1) * 2;     // qq tiles tj0, tj0+1

  v8f acc0 = vzero8();
  v8f acc1 = vzero8();

  for (int yc = 0; yc < 4096; yc += 128) {
    // stage: transpose 128x64 -> 64x128 (two threads per y-row)
    {
      const int y  = tid >> 1;
      const int d0 = (tid & 1) * 32;
      const bf16_t* qr = Q + (size_t)(yc + y) * 64 + d0;
      const bf16_t* kr = K + (size_t)(yc + y) * 64 + d0;
#pragma unroll
      for (int c = 0; c < 32; c += 8) {
        v8bf qv = *(const v8bf*)(qr + c);
        v8bf kv = *(const v8bf*)(kr + c);
#pragma unroll
        for (int e = 0; e < 8; ++e) {
          Qt[(d0 + c + e) * 128 + y] = qv[e];
          Kt[(d0 + c + e) * 128 + y] = kv[e];
        }
      }
    }
    __syncthreads();
#pragma unroll
    for (int y0 = 0; y0 < 128; y0 += 32) {
      v16bf af = load_frag_bf16(Kt, ti * 16,        128, y0, lane);  // A = K^T
      v16bf b0 = load_frag_bf16(Qt, tj0 * 16,       128, y0, lane);  // B = Q
      v16bf b1 = load_frag_bf16(Qt, (tj0 + 1) * 16, 128, y0, lane);
      acc0 = wmma_bf16(af, b0, acc0);
      acc1 = wmma_bf16(af, b1, acc1);
    }
    __syncthreads();
  }

  // deposit accumulators (each wave owns its tiles exclusively)
  {
    const int kkb = ti * 16 + ((lane >> 4) << 3);
    const int q0  = tj0 * 16 + (lane & 15);
#pragma unroll
    for (int e = 0; e < 8; ++e) {
      Z[(kkb + e) * 64 + q0]      = acc0[e];
      Z[(kkb + e) * 64 + q0 + 16] = acc1[e];
    }
  }
  __syncthreads();

  // softmax over qq (row-contiguous), scale 1/sqrt(64)=0.125
  if (tid < 64) {
    const float* row = Z + tid * 64;
    float mx = -3.0e38f;
#pragma unroll 8
    for (int c = 0; c < 64; ++c) mx = fmaxf(mx, row[c] * 0.125f);
    float sum = 0.0f;
#pragma unroll 8
    for (int c = 0; c < 64; ++c) sum += __expf(row[c] * 0.125f - mx);
    const float inv = 1.0f / sum;
    bf16_t* dst = attn + (size_t)bh * 4096 + tid * 64;
#pragma unroll 8
    for (int c = 0; c < 64; ++c)
      dst[c] = (bf16_t)(__expf(row[c] * 0.125f - mx) * inv);
  }
}

// ---------------------------------------------------------------------------
// z2[s][z] = sum_y A[y][z] * V[s][y]; written merged: z2m[b][s][h*64+z] (bf16)
// A (64x64) staged transposed in LDS; V rows feed A-fragments directly.
// ---------------------------------------------------------------------------
__global__ __launch_bounds__(256)
void av_kernel(const bf16_t* __restrict__ vh,
               const bf16_t* __restrict__ attn,
               bf16_t* __restrict__ z2m) {
  __shared__ __align__(16) bf16_t At[64 * 64];   // [z][y]
  const int tid  = threadIdx.x;
  const int lane = tid & 31;
  const int wid  = tid >> 5;
  const int bh   = blockIdx.y;
  const int m0   = blockIdx.x * 256 + wid * 32;
  const bf16_t* V = vh   + (size_t)bh * 4096 * 64;
  const bf16_t* A = attn + (size_t)bh * 4096;

  for (int idx = tid; idx < 4096; idx += 256) {
    const int y = idx >> 6, z = idx & 63;
    At[z * 64 + y] = A[idx];
  }
  __syncthreads();

  v8f acc[2][4];
#pragma unroll
  for (int i = 0; i < 2; ++i)
#pragma unroll
    for (int j = 0; j < 4; ++j) acc[i][j] = vzero8();

#pragma unroll
  for (int k0 = 0; k0 < 64; k0 += 32) {
    v16bf a0 = load_frag_bf16(V, m0,      64, k0, lane);
    v16bf a1 = load_frag_bf16(V, m0 + 16, 64, k0, lane);
#pragma unroll
    for (int j = 0; j < 4; ++j) {
      v16bf bf = load_frag_bf16(At, j * 16, 64, k0, lane);
      acc[0][j] = wmma_bf16(a0, bf, acc[0][j]);
      acc[1][j] = wmma_bf16(a1, bf, acc[1][j]);
    }
  }

  const int b = bh >> 4, h = bh & 15;
#pragma unroll
  for (int i = 0; i < 2; ++i) {
#pragma unroll
    for (int j = 0; j < 4; ++j) {
      const int col = h * 64 + j * 16 + (lane & 15);
#pragma unroll
      for (int e = 0; e < 8; ++e) {
        const int s = m0 + i * 16 + ((lane >> 4) << 3) + e;
        z2m[(size_t)(b * 4096 + s) * 1024 + col] = (bf16_t)acc[i][j][e];
      }
    }
  }
}

// ---------------------------------------------------------------------------
// Output projection: out (fp32) = z2m @ Wo^T  (32768 x 1024 x 1024)
// ---------------------------------------------------------------------------
__global__ __launch_bounds__(256)
void out_proj_kernel(const bf16_t* __restrict__ z2m,
                     const bf16_t* __restrict__ wo,
                     float* __restrict__ out) {
  const int tid  = threadIdx.x;
  const int lane = tid & 31;
  const int wid  = tid >> 5;
  const int n0   = blockIdx.x * 128 + (wid >> 2) * 64;
  const int m0   = blockIdx.y * 128 + (wid & 3) * 32;

  v8f acc[2][4];
#pragma unroll
  for (int i = 0; i < 2; ++i)
#pragma unroll
    for (int j = 0; j < 4; ++j) acc[i][j] = vzero8();

  for (int k0 = 0; k0 < 1024; k0 += 32) {
    if (k0 + 32 < 1024)
      __builtin_prefetch(z2m + (size_t)(m0 + (lane & 15)) * 1024 + k0 + 32, 0, 1);
    v16bf a0 = load_frag_bf16(z2m, m0,      1024, k0, lane);
    v16bf a1 = load_frag_bf16(z2m, m0 + 16, 1024, k0, lane);
    v16bf b[4];
#pragma unroll
    for (int j = 0; j < 4; ++j) b[j] = load_frag_bf16(wo, n0 + j * 16, 1024, k0, lane);
#pragma unroll
    for (int j = 0; j < 4; ++j) {
      acc[0][j] = wmma_bf16(a0, b[j], acc[0][j]);
      acc[1][j] = wmma_bf16(a1, b[j], acc[1][j]);
    }
  }

#pragma unroll
  for (int i = 0; i < 2; ++i) {
#pragma unroll
    for (int j = 0; j < 4; ++j) {
      const int n = n0 + j * 16 + (lane & 15);
#pragma unroll
      for (int e = 0; e < 8; ++e) {
        const int m = m0 + i * 16 + ((lane >> 4) << 3) + e;
        out[(size_t)m * 1024 + n] = acc[i][j][e];
      }
    }
  }
}

// ---------------------------------------------------------------------------
extern "C" void kernel_launch(void* const* d_in, const int* in_sizes, int n_in,
                              void* d_out, int out_size, void* d_ws, size_t ws_size,
                              hipStream_t stream) {
  (void)in_sizes; (void)n_in; (void)out_size; (void)ws_size;
  const float* x  = (const float*)d_in[0];
  const float* Wq = (const float*)d_in[1];
  const float* Wk = (const float*)d_in[2];
  const float* Wv = (const float*)d_in[3];
  const float* Wo = (const float*)d_in[4];
  float* out = (float*)d_out;

  char* ws = (char*)d_ws;
  size_t off = 0;
  auto walloc = [&](size_t bytes) {
    char* p = ws + off;
    off = (off + bytes + 255) & ~(size_t)255;
    return p;
  };

  const size_t D2   = 1024ull * 1024ull;           // weight elements
  const size_t XN   = 8ull * 4096ull * 1024ull;    // x elements
  const size_t QKVN = 128ull * 4096ull * 64ull;    // head-split q/k/v elements
  bf16_t* wqb  = (bf16_t*)walloc(D2 * 2);
  bf16_t* wkb  = (bf16_t*)walloc(D2 * 2);
  bf16_t* wvb  = (bf16_t*)walloc(D2 * 2);
  bf16_t* wob  = (bf16_t*)walloc(D2 * 2);
  bf16_t* xb   = (bf16_t*)walloc(XN * 2);
  bf16_t* qhb  = (bf16_t*)walloc(QKVN * 2);
  bf16_t* khb  = (bf16_t*)walloc(QKVN * 2);
  bf16_t* vhb  = (bf16_t*)walloc(QKVN * 2);
  bf16_t* attn = (bf16_t*)walloc(128ull * 64 * 64 * 2);
  bf16_t* z2m  = (bf16_t*)walloc(32768ull * 1024 * 2);

  convert_f32_bf16_vec8_kernel<<<512, 256, 0, stream>>>(Wq, wqb, (int)(D2 / 8));
  convert_f32_bf16_vec8_kernel<<<512, 256, 0, stream>>>(Wk, wkb, (int)(D2 / 8));
  convert_f32_bf16_vec8_kernel<<<512, 256, 0, stream>>>(Wv, wvb, (int)(D2 / 8));
  convert_f32_bf16_vec8_kernel<<<512, 256, 0, stream>>>(Wo, wob, (int)(D2 / 8));
  convert_f32_bf16_vec8_kernel<<<4096, 256, 0, stream>>>(x, xb, (int)(XN / 8));

  qkv_proj_kernel<<<dim3(24, 256), 256, 0, stream>>>(xb, wqb, wkb, wvb, qhb, khb, vhb);
  scores_softmax_kernel<<<128, 256, 0, stream>>>(qhb, khb, attn);
  av_kernel<<<dim3(16, 128), 256, 0, stream>>>(vhb, attn, z2m);
  out_proj_kernel<<<dim3(8, 256), 256, 0, stream>>>(z2m, wob, out);
}
